// ReprogrammingLayer_16569983828728
// MI455X (gfx1250) — matrix operand order repaired and checked by reading, not compiled, not run on hardware
//
#include <hip/hip_runtime.h>

// ---------------------------------------------------------------------------
// Types for CDNA5 (gfx1250) WMMA, wave32
// ---------------------------------------------------------------------------
typedef __bf16 bf16_t;
typedef bf16_t v4bf  __attribute__((ext_vector_type(4)));
typedef bf16_t v8bf  __attribute__((ext_vector_type(8)));
typedef bf16_t v16bf __attribute__((ext_vector_type(16)));
typedef float  v8f   __attribute__((ext_vector_type(8)));
typedef unsigned int u32x4 __attribute__((ext_vector_type(4)));
typedef int          i32x4 __attribute__((ext_vector_type(4)));
typedef int          i32x8 __attribute__((ext_vector_type(8)));

#if defined(__has_builtin)
#if __has_builtin(__builtin_amdgcn_tensor_load_to_lds) && \
    __has_builtin(__builtin_amdgcn_s_wait_tensorcnt)
#define ATHENA_HAVE_TDM 1
#endif
#endif

#define DEVINL static __device__ __forceinline__

DEVINL v16bf cat8(v8bf lo, v8bf hi) {
  return __builtin_shufflevector(lo, hi, 0,1,2,3,4,5,6,7,8,9,10,11,12,13,14,15);
}
DEVINL v8f zero8() {
  v8f z = {0.f,0.f,0.f,0.f,0.f,0.f,0.f,0.f};
  return z;
}
DEVINL v8f wmma_bf16(v16bf a, v16bf b, v8f c) {
  // D = A(16x32 bf16) x B(32x16 bf16) + C(16x16 f32)
  return __builtin_amdgcn_wmma_f32_16x16x32_bf16(false, a, false, b,
                                                 (short)0, c, false, false);
}

#if ATHENA_HAVE_TDM
// Issue a TDM 2D tile load: rows x (128B per row) from global (row stride
// Ktotal*2 bytes) into LDS at lds_off, padded +16B per row (LDS stride 144B).
// D# layout per CDNA5 ISA ch.8 (groups 0/1; groups 2/3 zero => <=2D tensor).
DEVINL void tdm_load_w_tile(unsigned lds_off, unsigned long long gaddr,
                            int Ktotal, int Ntotal) {
  u32x4 g0 = {
      1u,                                            // count=1, user D#
      lds_off,                                       // lds_addr (bytes)
      (unsigned)(gaddr & 0xffffffffull),             // global_addr[31:0]
      (unsigned)((gaddr >> 32) & 0x01ffffffull)      // global_addr[56:32]
          | (2u << 30)                               // type=2 ("image")
  };
  const unsigned kt = (unsigned)Ktotal, nt = (unsigned)Ntotal;
  i32x8 g1 = {
      (int)((1u << 16)        // data_size = 1 -> 2 bytes (bf16)
            | (1u << 20)      // pad_enable
            | (4u << 22)      // pad_interval code 4 -> every 32 DW (128B)
            | (3u << 25)),    // pad_amount  code 3 -> 4 DW (16B)
      (int)((kt & 0xffffu) << 16),                        // tensor_dim0 lo
      (int)(((kt >> 16) & 0xffffu) | ((nt & 0xffffu) << 16)),  // dim0 hi|dim1 lo
      (int)(((nt >> 16) & 0xffffu) | (64u << 16)),        // dim1 hi | tile_dim0=64
      (int)(128u),                                        // tile_dim1=128, tile_dim2=0
      (int)kt,                                            // tensor_dim0_stride lo
      0, 0                                                // stride hi, dim1_stride
  };
  i32x4 gz = {0, 0, 0, 0};
#if __clang_major__ >= 23
  i32x8 gz8 = {0, 0, 0, 0, 0, 0, 0, 0};
  __builtin_amdgcn_tensor_load_to_lds(g0, g1, gz, gz, gz8, 0);
#else
  __builtin_amdgcn_tensor_load_to_lds(g0, g1, gz, gz, 0);
#endif
}
#endif

// Problem constants (match reference)
constexpr int kDModel = 1024;
constexpr int kHeads  = 8;
constexpr int kDK     = 128;   // d per head
constexpr int kDLLM   = 2048;
constexpr int kDInner = 1024;  // kHeads * kDK
constexpr int kB      = 8;
constexpr int kL      = 128;
constexpr int kS      = 256;

// ---------------------------------------------------------------------------
// f32 -> bf16 conversion (weights)
// ---------------------------------------------------------------------------
__global__ __launch_bounds__(256) void cvt_f32_bf16(const float* __restrict__ s,
                                                    bf16_t* __restrict__ d, int n4) {
  int i = blockIdx.x * blockDim.x + threadIdx.x;
  if (i < n4) {
    float4 v = ((const float4*)s)[i];
    v4bf o = {(bf16_t)v.x, (bf16_t)v.y, (bf16_t)v.z, (bf16_t)v.w};
    ((v4bf*)d)[i] = o;
  }
}

// ---------------------------------------------------------------------------
// Tiled WMMA GEMM:  Out[M,N] = X[M,K] @ W[N,K]^T + bias[N]
//   X : TX (float -> converted to bf16 while staging, or bf16)
//   W : bf16 row-major (torch Linear layout, K-contiguous -> perfect B operand),
//       tile DMA'd into LDS by the Tensor Data Mover (TENSORcnt) when available.
//   Tile: 128x128 per WG (256 thr = 8 waves), K-step 64.
//   Wave (wm,wn) owns 32x64 -> 2x4 grid of 16x16 WMMA tiles (64 acc VGPRs).
// ---------------------------------------------------------------------------
template <typename TX, typename TO>
__global__ __launch_bounds__(256)
void gemm_wmma_bf16(const TX* __restrict__ X, const bf16_t* __restrict__ W,
                    const float* __restrict__ bias, TO* __restrict__ Out,
                    int Ntotal, int Ktotal) {
  __shared__ bf16_t sX[128][72];  // 128B data + 16B pad -> conflict-free A reads
  __shared__ bf16_t sW[128][72];  // TDM writes with pad_interval/pad_amount

  const int t    = threadIdx.x;
  const int w    = t >> 5;
  const int lane = t & 31;
  const int g    = lane >> 4;     // half-wave group (K-offset select)
  const int mn   = lane & 15;     // M row (A) / N col (B,D)
  const int wm   = w & 3;         // wave's 32-row quadrant
  const int wn   = w >> 2;        // wave's 64-col half
  const long m0  = (long)blockIdx.y * 128;
  const long n0  = (long)blockIdx.x * 128;

#if ATHENA_HAVE_TDM
  const unsigned lds_w = (unsigned)(size_t)(&sW[0][0]);
  const unsigned long long wbase =
      (unsigned long long)(size_t)W +
      (unsigned long long)(size_t)n0 * (unsigned)Ktotal * 2ull;
#endif

  v8f acc[2][4];
#pragma unroll
  for (int mi = 0; mi < 2; ++mi)
#pragma unroll
    for (int ni = 0; ni < 4; ++ni) acc[mi][ni] = zero8();

  for (int k0 = 0; k0 < Ktotal; k0 += 64) {
    __syncthreads();  // all waves done reading the previous tiles
#if ATHENA_HAVE_TDM
    if (w == 0)       // one TDM issue per WG; EXEC ignored by tensor ops
      tdm_load_w_tile(lds_w, wbase + (unsigned long long)k0 * 2ull,
                      Ktotal, Ntotal);
#endif
    // ---- stage 128x64 X tile (fused f32->bf16 when TX==float) ----------
#pragma unroll
    for (int it = 0; it < 4; ++it) {
      const int idx = t + it * 256;      // 1024 chunks of 8 elems
      const int row = idx >> 3;
      const int c8  = (idx & 7) << 3;
      const TX* xs = X + (size_t)(m0 + row) * Ktotal + k0 + c8;
      v8bf xv;
      if constexpr (__is_same(TX, float)) {
        float f[8];
        *(float4*)(f)     = *(const float4*)(xs);
        *(float4*)(f + 4) = *(const float4*)(xs + 4);
#pragma unroll
        for (int i = 0; i < 8; ++i) xv[i] = (bf16_t)f[i];
      } else {
        xv = *(const v8bf*)xs;
      }
      *(v8bf*)&sX[row][c8] = xv;
      if (k0 + 64 < Ktotal) __builtin_prefetch(xs + 64, 0, 1);
#if !ATHENA_HAVE_TDM
      const bf16_t* wsrc = W + (size_t)(n0 + row) * Ktotal + k0 + c8;
      *(v8bf*)&sW[row][c8] = *(const v8bf*)wsrc;
      if (k0 + 64 < Ktotal) __builtin_prefetch(wsrc + 64, 0, 1);
#endif
    }
#if ATHENA_HAVE_TDM
    if (w == 0) __builtin_amdgcn_s_wait_tensorcnt(0);
#endif
    __syncthreads();
    // ---- 16 WMMAs per wave ---------------------------------------------
#pragma unroll
    for (int kk = 0; kk < 2; ++kk) {
      v16bf a[2], b[4];
#pragma unroll
      for (int mi = 0; mi < 2; ++mi) {
        const int r = wm * 32 + mi * 16 + mn;
        a[mi] = cat8(*(const v8bf*)&sX[r][kk * 32 + 8 * g],
                     *(const v8bf*)&sX[r][kk * 32 + 8 * g + 16]);
      }
#pragma unroll
      for (int ni = 0; ni < 4; ++ni) {
        const int r = wn * 64 + ni * 16 + mn;
        b[ni] = cat8(*(const v8bf*)&sW[r][kk * 32 + 8 * g],
                     *(const v8bf*)&sW[r][kk * 32 + 8 * g + 16]);
      }
#pragma unroll
      for (int mi = 0; mi < 2; ++mi)
#pragma unroll
        for (int ni = 0; ni < 4; ++ni)
          acc[mi][ni] = wmma_bf16(a[mi], b[ni], acc[mi][ni]);
    }
  }

  // ---- epilogue: bias add + store -------------------------------------
#pragma unroll
  for (int mi = 0; mi < 2; ++mi) {
#pragma unroll
    for (int ni = 0; ni < 4; ++ni) {
      const long ncol = n0 + wn * 64 + ni * 16 + mn;
      const float bv = bias[ncol];
#pragma unroll
      for (int r = 0; r < 8; ++r) {
        const long mrow = m0 + wm * 32 + mi * 16 + r + 8 * g;  // D: M = r + 8g
        const float v = acc[mi][ni][r] + bv;
        Out[(size_t)mrow * Ntotal + ncol] = (TO)v;
      }
    }
  }
}

// ---------------------------------------------------------------------------
// Attention: one WG (8 waves) per (h,l).  Q(8x128) padded to 16 rows.
// scores = Q K^T * 1/sqrt(128); softmax over S=256; ctx = A V.
// K/V are bf16 and L2-resident (128 MB < 192 MB L2).
// ---------------------------------------------------------------------------
__global__ __launch_bounds__(256)
void attention_wmma(const bf16_t* __restrict__ Qb, const bf16_t* __restrict__ Kb,
                    const bf16_t* __restrict__ Vb, bf16_t* __restrict__ Ctx) {
  __shared__ bf16_t sQ[16][136];        // padded: conflict-free A reads
  __shared__ bf16_t sVt[128][264];      // V transposed: [e][s], s-contiguous B reads
  __shared__ float  sS[16][257];        // f32 scores / exp scratch
  __shared__ bf16_t sP[16][264];        // softmax probs, A-operand layout (row-major)
  __shared__ float  sRm[16][17];
  __shared__ float  sRs[16][17];

  const int hl = blockIdx.x;
  const int h  = hl >> 7;               // / kL
  const int l  = hl & 127;              // % kL
  const int t    = threadIdx.x;
  const int w    = t >> 5;
  const int lane = t & 31;
  const int g    = lane >> 4;
  const int mn   = lane & 15;

  // ---- stage Q (zero-pad rows 8..15) ----------------------------------
  {
    const int row = t >> 4;             // 16 rows
    const int c8  = (t & 15) << 3;      // 16 chunks of 8
    v8bf qv;
#pragma unroll
    for (int i = 0; i < 8; ++i) qv[i] = (bf16_t)0.0f;
    if (row < kB)
      qv = *(const v8bf*)(Qb + ((size_t)(row * kL + l)) * kDInner + h * kDK + c8);
    *(v8bf*)&sQ[row][c8] = qv;
  }
  // ---- stage V transposed: sVt[e][s] = V[s*L+l][h*128+e] --------------
  for (int idx = t; idx < kS * 16; idx += 256) {   // 256 rows x 16 chunks
    const int r  = idx >> 4;
    const int c8 = (idx & 15) << 3;
    v8bf v = *(const v8bf*)(Vb + ((size_t)(r * kL + l)) * kDInner + h * kDK + c8);
#pragma unroll
    for (int j = 0; j < 8; ++j) sVt[c8 + j][r] = v[j];
  }
  __syncthreads();

  // ---- scores: wave w does s-tiles 2w, 2w+1 ---------------------------
  const float scale = 0.08838834764831845f;  // 1/sqrt(128)
#pragma unroll
  for (int sti = 0; sti < 2; ++sti) {
    const int st = w * 2 + sti;
    v8f acc = zero8();
#pragma unroll
    for (int kc = 0; kc < 4; ++kc) {     // E = 128 = 4 * 32
      v16bf a = cat8(*(const v8bf*)&sQ[mn][kc * 32 + 8 * g],
                     *(const v8bf*)&sQ[mn][kc * 32 + 8 * g + 16]);
      const bf16_t* kp = Kb + ((size_t)((st * 16 + mn) * kL + l)) * kDInner +
                         h * kDK + kc * 32 + 8 * g;
      v16bf b = cat8(*(const v8bf*)kp, *(const v8bf*)(kp + 16));
      acc = wmma_bf16(a, b, acc);
    }
#pragma unroll
    for (int r = 0; r < 8; ++r) sS[r + 8 * g][st * 16 + mn] = acc[r] * scale;
  }
  __syncthreads();

  // ---- softmax over 256, 16 threads per row ---------------------------
  {
    const int row = t >> 4;
    const int seg = t & 15;
    float m = -3.4e38f;
#pragma unroll
    for (int i = 0; i < 16; ++i) m = fmaxf(m, sS[row][seg * 16 + i]);
    sRm[row][seg] = m;
    __syncthreads();
    if (seg == 0) {
      float mm = sRm[row][0];
#pragma unroll
      for (int i = 1; i < 16; ++i) mm = fmaxf(mm, sRm[row][i]);
      sRm[row][16] = mm;
    }
    __syncthreads();
    const float mm = sRm[row][16];
    float ssum = 0.f;
#pragma unroll
    for (int i = 0; i < 16; ++i) {
      float e = __expf(sS[row][seg * 16 + i] - mm);
      sS[row][seg * 16 + i] = e;
      ssum += e;
    }
    sRs[row][seg] = ssum;
    __syncthreads();
    if (seg == 0) {
      float s = 0.f;
#pragma unroll
      for (int i = 0; i < 16; ++i) s += sRs[row][i];
      sRs[row][16] = 1.0f / s;
    }
    __syncthreads();
    const float inv = sRs[row][16];
#pragma unroll
    for (int i = 0; i < 16; ++i)
      sP[row][seg * 16 + i] = (bf16_t)(sS[row][seg * 16 + i] * inv);
  }
  __syncthreads();

  // ---- ctx = A V : wave w owns e-tile w -------------------------------
  v8f acc = zero8();
#pragma unroll
  for (int kc = 0; kc < 8; ++kc) {       // S = 256 = 8 * 32
    v16bf a = cat8(*(const v8bf*)&sP[mn][kc * 32 + 8 * g],
                   *(const v8bf*)&sP[mn][kc * 32 + 8 * g + 16]);
    const bf16_t* vp = &sVt[w * 16 + mn][kc * 32 + 8 * g];
    v16bf b = cat8(*(const v8bf*)vp, *(const v8bf*)(vp + 16));
    acc = wmma_bf16(a, b, acc);
  }
  if (g == 0) {                          // rows M = r (0..7) are the valid b's
#pragma unroll
    for (int r = 0; r < 8; ++r)
      Ctx[((size_t)(r * kL + l)) * kDInner + h * kDK + w * 16 + mn] = (bf16_t)acc[r];
  }
}

// ---------------------------------------------------------------------------
// Host-side launch
// ---------------------------------------------------------------------------
extern "C" void kernel_launch(void* const* d_in, const int* in_sizes, int n_in,
                              void* d_out, int out_size, void* d_ws, size_t ws_size,
                              hipStream_t stream) {
  (void)in_sizes; (void)n_in; (void)out_size; (void)ws_size;
  const float* llm = (const float*)d_in[0];
  const float* src = (const float*)d_in[1];
  const float* val = (const float*)d_in[2];
  const float* Wq  = (const float*)d_in[3];
  const float* bq  = (const float*)d_in[4];
  const float* Wk  = (const float*)d_in[5];
  const float* bk  = (const float*)d_in[6];
  const float* Wv  = (const float*)d_in[7];
  const float* bv  = (const float*)d_in[8];
  const float* Wo  = (const float*)d_in[9];
  const float* bo  = (const float*)d_in[10];
  float* out = (float*)d_out;

  // Workspace layout (bytes)
  char* ws = (char*)d_ws;
  constexpr size_t MB = 1024ull * 1024ull;
  bf16_t* Wq_b = (bf16_t*)(ws + 0);        // 2 MB
  bf16_t* Wk_b = (bf16_t*)(ws + 2 * MB);   // 4 MB
  bf16_t* Wv_b = (bf16_t*)(ws + 6 * MB);   // 4 MB
  bf16_t* Wo_b = (bf16_t*)(ws + 10 * MB);  // 4 MB
  bf16_t* Qb   = (bf16_t*)(ws + 14 * MB);  // 2 MB  (B*L, 1024)
  bf16_t* Kb   = (bf16_t*)(ws + 16 * MB);  // 64 MB (S*L, 1024)
  bf16_t* Vb   = (bf16_t*)(ws + 80 * MB);  // 64 MB
  bf16_t* Ctx  = (bf16_t*)(ws + 144 * MB); // 2 MB

  // 1) weights -> bf16
  {
    int n4;
    n4 = kDInner * kDModel / 4;
    cvt_f32_bf16<<<n4 / 256, 256, 0, stream>>>(Wq, Wq_b, n4);
    n4 = kDInner * kDLLM / 4;
    cvt_f32_bf16<<<n4 / 256, 256, 0, stream>>>(Wk, Wk_b, n4);
    cvt_f32_bf16<<<n4 / 256, 256, 0, stream>>>(Wv, Wv_b, n4);
    n4 = kDLLM * kDInner / 4;
    cvt_f32_bf16<<<n4 / 256, 256, 0, stream>>>(Wo, Wo_b, n4);
  }

  // 2) Q = llm @ Wq^T + bq          (M=1024, N=1024, K=1024)
  gemm_wmma_bf16<float, bf16_t>
      <<<dim3(kDInner / 128, (kB * kL) / 128), 256, 0, stream>>>(
          llm, Wq_b, bq, Qb, kDInner, kDModel);

  // 3) K = src @ Wk^T + bk          (M=32768, N=1024, K=2048)
  gemm_wmma_bf16<float, bf16_t>
      <<<dim3(kDInner / 128, (kS * kL) / 128), 256, 0, stream>>>(
          src, Wk_b, bk, Kb, kDInner, kDLLM);

  // 4) V = val @ Wv^T + bv
  gemm_wmma_bf16<float, bf16_t>
      <<<dim3(kDInner / 128, (kS * kL) / 128), 256, 0, stream>>>(
          val, Wv_b, bv, Vb, kDInner, kDLLM);

  // 5) attention per (h,l)
  attention_wmma<<<kHeads * kL, 256, 0, stream>>>(Qb, Kb, Vb, Ctx);

  // 6) out = ctx @ Wo^T + bo        (M=1024, N=2048, K=1024) -> f32 d_out
  gemm_wmma_bf16<bf16_t, float>
      <<<dim3(kDLLM / 128, (kB * kL) / 128), 256, 0, stream>>>(
          Ctx, Wo_b, bo, out, kDLLM, kDInner);
}
